// ChannelTR_22084721836076
// MI455X (gfx1250) — compile-verified, matrix operand order checked by reading, not run on hardware
//
#include <hip/hip_runtime.h>

typedef __bf16 bf16;
typedef __attribute__((ext_vector_type(16))) __bf16 v16bf;
typedef __attribute__((ext_vector_type(8)))  __bf16 v8bf;
typedef __attribute__((ext_vector_type(4)))  __bf16 v4bf;
typedef __attribute__((ext_vector_type(8)))  float  v8f;

constexpr int XC    = 256;    // channels (C)
constexpr int NPIX  = 4096;   // H*W
constexpr int DIMH  = 512;    // per-head dim (N/NH)
constexpr int KC    = 64;     // K chunk staged in LDS
constexpr int PITCH = 72;     // LDS row pitch (bf16 elems) = 144B, multiple of 16B

// ---- WMMA fragment assembly from LDS (A-layout; B operand = B^T in same layout) ----
__device__ __forceinline__ v16bf load_frag(const bf16* rowbase, int ks, int lane) {
  const bf16* p = rowbase + ks + ((lane >> 4) << 3);
  v8bf lo = *(const v8bf*)(p);        // ds_load_b128
  v8bf hi = *(const v8bf*)(p + 16);   // ds_load_b128
  return __builtin_shufflevector(lo, hi, 0,1,2,3,4,5,6,7,8,9,10,11,12,13,14,15);
}

__device__ __forceinline__ v8f wmma_bf16(v16bf a, v16bf b, v8f c) {
  return __builtin_amdgcn_wmma_f32_16x16x32_bf16(false, a, false, b, (short)0, c, false, false);
}

__device__ __forceinline__ float wave_min32(float v) {
  for (int off = 16; off > 0; off >>= 1) v = fminf(v, __shfl_xor(v, off, 32));
  return v;
}
__device__ __forceinline__ float wave_sum32(float v) {
  for (int off = 16; off > 0; off >>= 1) v += __shfl_xor(v, off, 32);
  return v;
}

// ---- CDNA5 async global->LDS copy (bf16 data, no conversion needed) ----
// VDST = per-lane LDS byte address, VADDR = 64-bit global address; ASYNCcnt tracked.
__device__ __forceinline__ void async_copy_b128(const void* gsrc, void* lds_dst) {
  unsigned int  ldsa = (unsigned int)(unsigned long long)lds_dst; // flat addr[31:0] == LDS offset
  unsigned long long ga = (unsigned long long)gsrc;
  asm volatile("global_load_async_to_lds_b128 %0, %1, off"
               :: "v"(ldsa), "v"(ga) : "memory");
}
__device__ __forceinline__ void async_wait_all() {
  asm volatile("s_wait_asynccnt 0x0" ::: "memory");
}

// ============ Kernel 0: XT[bh][d][k] = bf16(x[b,k,h,d]) (LDS-tiled transpose) ============
__global__ __launch_bounds__(256) void xpose_kernel(const float* __restrict__ x,
                                                    bf16* __restrict__ xt) {
  __shared__ __align__(16) bf16 tile[64 * PITCH];
  const int tid = threadIdx.x;
  const int kt = blockIdx.x, dt = blockIdx.y, bh = blockIdx.z;
  const int bb = bh >> 3, hh = bh & 7;
  const float* xb = x + (size_t)bb * XC * NPIX + (size_t)hh * DIMH;

  const int rr = tid >> 4;            // 0..15
  const int cc = (tid & 15) << 2;     // 0..60
  for (int i = 0; i < 4; ++i) {
    int k = rr + 16 * i;
    float4 v = *(const float4*)(xb + (size_t)(kt * 64 + k) * NPIX + dt * 64 + cc);
    v4bf pk = {(bf16)v.x, (bf16)v.y, (bf16)v.z, (bf16)v.w};
    *(v4bf*)(tile + k * PITCH + cc) = pk;
  }
  __syncthreads();
  for (int i = 0; i < 4; ++i) {
    int d = rr + 16 * i;
    v4bf pk = { tile[(cc + 0) * PITCH + d], tile[(cc + 1) * PITCH + d],
                tile[(cc + 2) * PITCH + d], tile[(cc + 3) * PITCH + d] };
    *(v4bf*)(xt + ((size_t)bh * DIMH + dt * 64 + d) * XC + kt * 64 + cc) = pk;
  }
}

// ============ Kernel 1: E = X·X^T (WMMA), softmax(-E) -> attn (bf16) ============
// grid(2 rowblocks of 128, 128 bh), 512 threads = 16 waves; wave w owns N-cols [16w,16w+16)
// Dynamic LDS: max(chunk 256x72 bf16 = 36864B, E 128x256 f32 = 131072B) = 131072B
__global__ __launch_bounds__(512) void attn_kernel(const float* __restrict__ x,
                                                   bf16* __restrict__ attn) {
  extern __shared__ __align__(16) char dynsmem[];
  bf16*  xbuf = (bf16*)dynsmem;       // 256 x KC bf16 chunk
  float* Ebuf = (float*)dynsmem;      // 128 x 256 f32

  const int tid = threadIdx.x, lane = tid & 31, wave = tid >> 5;
  const int bh = blockIdx.y, bb = bh >> 3, hh = bh & 7;
  const int c0 = blockIdx.x * 128;
  const float* xb = x + (size_t)bb * XC * NPIX + (size_t)hh * DIMH;

  v8f acc[8] = {};
  const int r0 = tid >> 4;            // 0..31 (global-load row group)
  const int cv = (tid & 15) << 2;     // 0..60 (float4 col)

  float4 stg[8];
  for (int i = 0; i < 8; ++i)         // prefetch chunk 0
    stg[i] = *(const float4*)(xb + (size_t)(r0 + 32 * i) * NPIX + cv);

  for (int ch = 0; ch < DIMH / KC; ++ch) {
    __syncthreads();                  // prior compute done before overwrite
    for (int i = 0; i < 8; ++i) {     // f32 -> packed bf16 -> LDS
      v4bf pk = {(bf16)stg[i].x, (bf16)stg[i].y, (bf16)stg[i].z, (bf16)stg[i].w};
      *(v4bf*)(xbuf + (r0 + 32 * i) * PITCH + cv) = pk;
    }
    if (ch + 1 < DIMH / KC) {         // prefetch next chunk (overlaps compute)
      const float* xn = xb + (ch + 1) * KC;
      for (int i = 0; i < 8; ++i)
        stg[i] = *(const float4*)(xn + (size_t)(r0 + 32 * i) * NPIX + cv);
    }
    __syncthreads();
    const int n0 = wave * 16;
    for (int ks = 0; ks < KC; ks += 32) {
      v16bf bf = load_frag(xbuf + (n0 + (lane & 15)) * PITCH, ks, lane);     // X rows k
      for (int m = 0; m < 8; ++m) {
        v16bf af = load_frag(xbuf + (c0 + m * 16 + (lane & 15)) * PITCH, ks, lane); // X rows c
        acc[m] = wmma_bf16(af, bf, acc[m]);
      }
    }
  }

  __syncthreads();
  const int n0 = wave * 16;
  for (int m = 0; m < 8; ++m)         // spill E tiles (C/D layout: M=r+8*lg, N=lane&15)
    for (int r = 0; r < 8; ++r)
      Ebuf[(m * 16 + ((lane >> 4) << 3) + r) * XC + n0 + (lane & 15)] = acc[m][r];
  __syncthreads();

  // softmax(-E) per row: attn = exp(rowmin(E) - E) / sum   (wave handles 8 rows)
  for (int i = 0; i < 8; ++i) {
    const int lr = wave * 8 + i;
    const float* er = Ebuf + lr * XC + lane * 8;
    float e[8];
    *(float4*)(e)     = *(const float4*)(er);
    *(float4*)(e + 4) = *(const float4*)(er + 4);
    float mn = e[0];
    for (int j = 1; j < 8; ++j) mn = fminf(mn, e[j]);
    mn = wave_min32(mn);
    float p[8], s = 0.f;
    for (int j = 0; j < 8; ++j) { p[j] = __expf(mn - e[j]); s += p[j]; }
    s = wave_sum32(s);
    const float inv = 1.f / s;
    v8bf pk;
    for (int j = 0; j < 8; ++j) pk[j] = (bf16)(p[j] * inv);
    *(v8bf*)(attn + ((size_t)bh * XC + c0 + lr) * XC + lane * 8) = pk;
  }
}

// ============ Kernel 2: out = gamma * (attn · X) + x  (WMMA, async LDS fills) ============
// grid(4 d-blocks, 128 bh), 512 threads = 16 waves in a 4x4 tile grid (64x32 per wave)
__global__ __launch_bounds__(512) void out_kernel(const float* __restrict__ x,
                                                  const bf16* __restrict__ attn,
                                                  const bf16* __restrict__ xt,
                                                  const float* __restrict__ gammap,
                                                  float* __restrict__ out) {
  __shared__ __align__(16) bf16 smem[(XC + 128) * PITCH];  // 55296B
  bf16* aBuf = smem;                 // attn rows 0..255, KC cols
  bf16* bBuf = smem + XC * PITCH;    // XT rows (d) 0..127, KC cols

  const int tid = threadIdx.x, lane = tid & 31, wave = tid >> 5;
  const int bh = blockIdx.y, bb = bh >> 3, hh = bh & 7;
  const int d0 = blockIdx.x * 128;
  const int mb = (wave >> 2) * 64, nb = (wave & 3) * 32;

  const bf16* aG = attn + (size_t)bh * XC * XC;
  const bf16* bG = xt + ((size_t)bh * DIMH + d0) * XC;

  v8f acc[4][2] = {};
  const int r0 = tid >> 3;           // 0..63
  const int cv = (tid & 7) << 3;     // 0..56

  for (int kc = 0; kc < XC; kc += KC) {
    __syncthreads();                 // all reads of previous chunk done
    // async bf16 copies straight into LDS (no VGPR staging, ASYNCcnt-tracked)
    for (int i = 0; i < 4; ++i) {    // attn chunk: 256 rows x 64
      int row = r0 + 64 * i;
      async_copy_b128(aG + (size_t)row * XC + kc + cv, aBuf + row * PITCH + cv);
    }
    for (int i = 0; i < 2; ++i) {    // XT chunk: 128 rows x 64
      int row = r0 + 64 * i;
      async_copy_b128(bG + (size_t)row * XC + kc + cv, bBuf + row * PITCH + cv);
    }
    async_wait_all();                // this wave's LDS writes landed
    __syncthreads();                 // everyone's writes visible
    for (int ks = 0; ks < KC; ks += 32) {
      v16bf bf0 = load_frag(bBuf + (nb +      (lane & 15)) * PITCH, ks, lane);
      v16bf bf1 = load_frag(bBuf + (nb + 16 + (lane & 15)) * PITCH, ks, lane);
      for (int m = 0; m < 4; ++m) {
        v16bf af = load_frag(aBuf + (mb + m * 16 + (lane & 15)) * PITCH, ks, lane);
        acc[m][0] = wmma_bf16(af, bf0, acc[m][0]);
        acc[m][1] = wmma_bf16(af, bf1, acc[m][1]);
      }
    }
  }

  const float g = gammap[0];
  const size_t base = (size_t)bb * XC * NPIX + (size_t)hh * DIMH + d0;
  for (int m = 0; m < 4; ++m)
    for (int n = 0; n < 2; ++n)
      for (int r = 0; r < 8; ++r) {
        const int c = mb + m * 16 + ((lane >> 4) << 3) + r;
        const int d = nb + n * 16 + (lane & 15);
        const size_t idx = base + (size_t)c * NPIX + d;
        out[idx] = g * acc[m][n][r] + x[idx];
      }
}

extern "C" void kernel_launch(void* const* d_in, const int* in_sizes, int n_in,
                              void* d_out, int out_size, void* d_ws, size_t ws_size,
                              hipStream_t stream) {
  const float* x     = (const float*)d_in[0];
  const float* gamma = (const float*)d_in[1];
  float* out = (float*)d_out;

  // workspace: attn (128*256*256 bf16 = 16MB) then XT (128*512*256 bf16 = 32MB)
  bf16* attnWS = (bf16*)d_ws;
  bf16* xtWS   = attnWS + (size_t)128 * XC * XC;

  // kernel1 uses 128KB dynamic LDS (above default static cap; WGP has 320KB)
  (void)hipFuncSetAttribute(reinterpret_cast<const void*>(&attn_kernel),
                            hipFuncAttributeMaxDynamicSharedMemorySize, 131072);

  xpose_kernel<<<dim3(4, 8, 128), 256, 0,      stream>>>(x, xtWS);
  attn_kernel <<<dim3(2, 128),    512, 131072, stream>>>(x, attnWS);
  out_kernel  <<<dim3(4, 128),    512, 0,      stream>>>(x, attnWS, xtWS, gamma, out);
}
